// GRU4RecUserModule_82995948027916
// MI455X (gfx1250) — matrix operand order repaired and checked by reading, not compiled, not run on hardware
//
#include <hip/hip_runtime.h>

typedef __attribute__((ext_vector_type(16))) __bf16 v16bf;
typedef __attribute__((ext_vector_type(2)))  __bf16 v2bf;
typedef __attribute__((ext_vector_type(8)))  float  v8f;

#define NSEG 16      // segments per workgroup = WMMA M
#define D_IN 64
#define H_GRU 128
#define MAXL 512

__device__ __forceinline__ unsigned short f2bf(float f) {
  union { __bf16 b; unsigned short s; } cv;
  cv.b = (__bf16)f;                       // hardware f32->bf16 (RNE)
  return cv.s;
}
__device__ __forceinline__ __bf16 f2bfh(float f) { return (__bf16)f; }

// pack two f32 -> packed bf16 dword (v_cvt_pk_bf16_f32 when available)
__device__ __forceinline__ unsigned int pk_bf16(float lo, float hi) {
#if __has_builtin(__builtin_amdgcn_cvt_pk_bf16_f32)
  union { v2bf v; unsigned int u; } cv;
  cv.v = __builtin_amdgcn_cvt_pk_bf16_f32(lo, hi);
  return cv.u;
#else
  return (unsigned)f2bf(lo) | ((unsigned)f2bf(hi) << 16);
#endif
}

#if __has_builtin(__builtin_amdgcn_rcpf)
__device__ __forceinline__ float fast_rcp(float x) { return __builtin_amdgcn_rcpf(x); }
#else
__device__ __forceinline__ float fast_rcp(float x) { return 1.0f / x; }
#endif

#if __has_builtin(__builtin_amdgcn_tanhf)
__device__ __forceinline__ float fast_tanh(float x) { return __builtin_amdgcn_tanhf(x); }
#else
__device__ __forceinline__ float fast_tanh(float x) {
  x = fminf(fmaxf(x, -30.0f), 30.0f);
  float e = __expf(-2.0f * x);
  return (1.0f - e) * fast_rcp(1.0f + e);
}
#endif
__device__ __forceinline__ float fast_sigmoid(float x) {
  return 0.5f * fast_tanh(0.5f * x) + 0.5f;
}

// A-fragment (16x32 bf16) from an LDS row: lane holds row M=lane%16,
// elements 0..7 -> K = kbase + 8*half + e, elements 8..15 -> K = kbase + 16 + 8*half + e
__device__ __forceinline__ v16bf ldsA(const unsigned short* row, int kbase, int half) {
  union { uint4 q[2]; v16bf v; } u;
  u.q[0] = *(const uint4*)(row + kbase + 8 * half);
  u.q[1] = *(const uint4*)(row + kbase + 16 + 8 * half);
  return u.v;
}

// B-fragment (32x16 bf16) from a global weight row (row = output column n):
// elements 0..7 -> K = kbase + 8*half + e, elements 8..15 -> K = kbase + 16 + 8*half + e
__device__ __forceinline__ v16bf loadB(const float* wrow, int kbase, int half) {
  v16bf v;
  const float* p = wrow + kbase + 8 * half;
#pragma unroll
  for (int e = 0; e < 16; ++e) v[e] = f2bfh(p[e < 8 ? e : e + 8]);
  return v;
}

__device__ __forceinline__ v8f wmma_bf16(v16bf a, v16bf b, v8f c) {
  return __builtin_amdgcn_wmma_f32_16x16x32_bf16(false, a, false, b, (short)0, c,
                                                 false, false);
}

__global__ void __launch_bounds__(256)
gru4rec_kernel(const float* __restrict__ x, const int* __restrict__ offsets,
               const float* __restrict__ W_ih, const float* __restrict__ W_hh,
               const float* __restrict__ W_dense, const float* __restrict__ b_dense,
               float* __restrict__ out, int T, int B)
{
  __shared__ unsigned short x_bf[2][NSEG][D_IN];    // ping-pong x_t (bf16)
  __shared__ unsigned short h_bf[2][NSEG][H_GRU];   // ping-pong h (bf16, WMMA A)
  __shared__ unsigned short hl_bf[NSEG][H_GRU];     // h at t = len-1
  __shared__ float out_s[NSEG][D_IN];
  __shared__ int   len_s[NSEG];
  __shared__ int   off_s[NSEG];
  __shared__ float scl_s[NSEG];

  const int tid  = threadIdx.x;
  const int wave = tid >> 5;
  const int lane = tid & 31;
  const int half = lane >> 4;
  const int nloc = lane & 15;

  // per-segment offsets / clipped lengths
  if (tid < NSEG) {
    int b = blockIdx.x * NSEG + tid;
    int off = 0, ln = 1;
    if (b < B) {
      off = offsets[b];
      int nxt = (b + 1 < B) ? offsets[b + 1] : T;
      ln = nxt - off;
      ln = ln < 1 ? 1 : (ln > MAXL ? MAXL : ln);
    }
    off_s[tid] = off;
    len_s[tid] = ln;
  }
  for (int i = tid; i < NSEG * H_GRU; i += 256)
    (&h_bf[0][0][0])[i] = 0;                        // zero h_bf[0] only
  __syncthreads();

  // ---- x staging descriptor: running byte offset, clamped at (T-1) ----
  const int sm = tid >> 4;                          // segment row 0..15
  const int sc = (tid & 15) * 4;                    // column 0..60 (step 4)
  long long boff = (long long)off_s[sm] * (D_IN * 4) + sc * 4;
  const long long bmax = (long long)(T - 1) * (D_IN * 4) + sc * 4;
  const char* xb = (const char*)x;

  // stage x_0 into buffer 0
  {
    long long a = boff < bmax ? boff : bmax;
    float4 xv = *(const float4*)(xb + a);
    uint2 pp;
    pp.x = pk_bf16(xv.x, xv.y);
    pp.y = pk_bf16(xv.z, xv.w);
    *(uint2*)&x_bf[0][sm][sc] = pp;
  }
  // software pipeline: issue load of x_1 now, consumed next iteration
  boff += D_IN * 4;
  float4 xnext;
  {
    long long a = boff < bmax ? boff : bmax;
    xnext = *(const float4*)(xb + a);
  }

  int maxlen = 0;
#pragma unroll
  for (int m = 0; m < NSEG; ++m) maxlen = max(maxlen, len_s[m]);
  int lenm[8];
#pragma unroll
  for (int vr = 0; vr < 8; ++vr) lenm[vr] = len_s[vr + 8 * half];

  // register-resident bf16 weight fragments: wave w owns hidden cols [16w,16w+16)
  // of each of the 3 gates. 18 v16bf frags = 144 VGPRs/lane.
  v16bf fih[3][2], fhh[3][4];
#pragma unroll
  for (int g = 0; g < 3; ++g) {
    int col = 128 * g + 16 * wave + nloc;           // row index of W_ih / W_hh
    const float* wi = W_ih + (size_t)col * D_IN;
    const float* wh = W_hh + (size_t)col * H_GRU;
#pragma unroll
    for (int kc = 0; kc < 2; ++kc) fih[g][kc] = loadB(wi, 32 * kc, half);
#pragma unroll
    for (int kc = 0; kc < 4; ++kc) fhh[g][kc] = loadB(wh, 32 * kc, half);
  }

  const int cl = 16 * wave + nloc;                  // hidden column this lane updates
  float hreg[8];                                    // f32 master h (registers)
  unsigned short hlast[8];                          // bf16 bits of h at t = len-1
#pragma unroll
  for (int vr = 0; vr < 8; ++vr) { hreg[vr] = 0.0f; hlast[vr] = 0; }

  __syncthreads();                                  // x_bf[0], h_bf[0] visible

  for (int t = 0; t < maxlen; ++t) {
    const int cur = t & 1, nxt = cur ^ 1;

    // store x_{t+1} (loaded one iteration ago) into the other buffer
    {
      uint2 pp;
      pp.x = pk_bf16(xnext.x, xnext.y);
      pp.y = pk_bf16(xnext.z, xnext.w);
      *(uint2*)&x_bf[nxt][sm][sc] = pp;
    }
    // issue load of x_{t+2}; waited one full iteration later
    boff += D_IN * 4;
    {
      long long a = boff < bmax ? boff : bmax;
      xnext = *(const float4*)(xb + a);
    }

    // A fragments for step t from the 'cur' buffers
    const unsigned short* xrow = &x_bf[cur][nloc][0];
    const unsigned short* hrow = &h_bf[cur][nloc][0];
    v16bf ax0 = ldsA(xrow,  0, half);
    v16bf ax1 = ldsA(xrow, 32, half);
    v16bf ah0 = ldsA(hrow,  0, half);
    v16bf ah1 = ldsA(hrow, 32, half);
    v16bf ah2 = ldsA(hrow, 64, half);
    v16bf ah3 = ldsA(hrow, 96, half);

    // 18 WMMAs; gh split into two independent 2-deep chains per gate (depth 2, not 4)
    v8f accS[3], accH[3];
#pragma unroll
    for (int g = 0; g < 3; ++g) {
      v8f s = {0,0,0,0,0,0,0,0};
      s = wmma_bf16(ax0, fih[g][0], s);
      s = wmma_bf16(ax1, fih[g][1], s);
      accS[g] = s;
      v8f hA = {0,0,0,0,0,0,0,0};
      v8f hB = {0,0,0,0,0,0,0,0};
      hA = wmma_bf16(ah0, fhh[g][0], hA);
      hB = wmma_bf16(ah2, fhh[g][2], hB);
      hA = wmma_bf16(ah1, fhh[g][1], hA);
      hB = wmma_bf16(ah3, fhh[g][3], hB);
      accH[g] = hA + hB;
    }

    // gate math in f32; each lane owns (rows vr+8*half, col cl)
#pragma unroll
    for (int vr = 0; vr < 8; ++vr) {
      float r = fast_sigmoid(accS[0][vr] + accH[0][vr]);
      float z = fast_sigmoid(accS[1][vr] + accH[1][vr]);
      float n = fast_tanh(accS[2][vr] + r * accH[2][vr]);
      float hnew = (1.0f - z) * n + z * hreg[vr];
      hreg[vr] = hnew;
      unsigned short hb = f2bf(hnew);
      h_bf[nxt][vr + 8 * half][cl] = hb;
      hlast[vr] = (t == lenm[vr] - 1) ? hb : hlast[vr];  // register snapshot
    }

    __syncthreads();   // single barrier: 'nxt' writes visible, 'cur' reads retired
  }

  // publish last-valid hidden states (bf16) once
#pragma unroll
  for (int vr = 0; vr < 8; ++vr)
    hl_bf[vr + 8 * half][cl] = hlast[vr];
  __syncthreads();

  // out = h_last . W_dense^T + b_dense   ([16x128]x[128x64]) via WMMA.
  // All 8 waves execute (EXEC all-1s as WMMA requires); waves 4..7 redundantly
  // compute the same tiles as waves 0..3 and store identical values.
  {
    int wsel = wave & 3;
    int col = 16 * wsel + nloc;                     // output dim 0..63
    const float* wd = W_dense + (size_t)col * H_GRU;
    v16bf bw0 = loadB(wd,  0, half), bw1 = loadB(wd, 32, half);
    v16bf bw2 = loadB(wd, 64, half), bw3 = loadB(wd, 96, half);
    const unsigned short* arow = &hl_bf[nloc][0];
    v16bf a0 = ldsA(arow,  0, half), a1 = ldsA(arow, 32, half);
    v16bf a2 = ldsA(arow, 64, half), a3 = ldsA(arow, 96, half);
    v8f acc = {0,0,0,0,0,0,0,0};
    acc = wmma_bf16(a0, bw0, acc);
    acc = wmma_bf16(a1, bw1, acc);
    acc = wmma_bf16(a2, bw2, acc);
    acc = wmma_bf16(a3, bw3, acc);
    float bias = b_dense[col];
#pragma unroll
    for (int vr = 0; vr < 8; ++vr)
      out_s[vr + 8 * half][col] = acc[vr] + bias;   // benign duplicate stores
  }
  __syncthreads();

  // row-wise L2 norm
  if (tid < NSEG) {
    float ss = 0.0f;
#pragma unroll
    for (int c = 0; c < D_IN; ++c) { float v = out_s[tid][c]; ss += v * v; }
    scl_s[tid] = 1.0f / fmaxf(sqrtf(ss), 1e-12f);
  }
  __syncthreads();

  {
    int idx = tid * 4, m = idx >> 6, c = idx & 63;
    int b = blockIdx.x * NSEG + m;
    if (b < B) {
      float s = scl_s[m];
      float4 o;
      o.x = out_s[m][c + 0] * s;
      o.y = out_s[m][c + 1] * s;
      o.z = out_s[m][c + 2] * s;
      o.w = out_s[m][c + 3] * s;
      *(float4*)(out + (size_t)b * D_IN + c) = o;
    }
  }
}

extern "C" void kernel_launch(void* const* d_in, const int* in_sizes, int n_in,
                              void* d_out, int out_size, void* d_ws, size_t ws_size,
                              hipStream_t stream) {
  const float* x       = (const float*)d_in[0];
  const int*   offsets = (const int*)  d_in[1];
  const float* W_ih    = (const float*)d_in[2];
  const float* W_hh    = (const float*)d_in[3];
  const float* W_dense = (const float*)d_in[4];
  const float* b_dense = (const float*)d_in[5];
  float* out = (float*)d_out;

  int T = in_sizes[0] / D_IN;
  int B = in_sizes[1];
  int groups = (B + NSEG - 1) / NSEG;

  gru4rec_kernel<<<dim3(groups), dim3(256), 0, stream>>>(
      x, offsets, W_ih, W_hh, W_dense, b_dense, out, T, B);
}
